// MimiEuclideanCodebook_45466523795677
// MI455X (gfx1250) — matrix operand order repaired and checked by reading, not compiled, not run on hardware
//
#include <hip/hip_runtime.h>

// MI455X / gfx1250, wave32.
// VQ nearest-neighbor: N=65536 x D=256 vs K=2048 codes.
// Distance GEMM via v_wmma_f32_16x16x32_bf16 with split-bf16 (hi/lo) operands
// (~fp32 accuracy, 3 bf16 WMMAs per 32-deep k-step vs 8 fp32 WMMAs).
// B tiles double-buffered in LDS; staged with GLOBAL_LOAD_ASYNC_TO_LDS_B128
// (ASYNCcnt) so tile ct+1 streams in under the WMMAs of tile ct.

typedef __attribute__((ext_vector_type(16))) __bf16 v16bf;
typedef __attribute__((ext_vector_type(8)))  float  v8f;
typedef int v4i __attribute__((vector_size(16)));   // matches builtin param type

#define KCODES 2048
#define DDIM   256
#define EPSV   1e-5f
#define NCT    (KCODES / 16)   // 128 code tiles
#define NDT    (DDIM / 32)     // 8 d-steps of 32

#define AS1 __attribute__((address_space(1)))
#define AS3 __attribute__((address_space(3)))

#if defined(__HIP_DEVICE_COMPILE__) && \
    __has_builtin(__builtin_amdgcn_global_load_async_to_lds_b128)
#define HAVE_ASYNC_LDS 1
#endif

// Workspace layout (dwords):
//   [0, 524288)             embed f32           (2 MB)
//   [524288, 526336)        e2 (|embed|^2)      (8 KB)
//   [526336, 1050624)       packed bf16 hi/lo codebook in WMMA-B order (2 MB)

static __device__ inline unsigned pack2(__bf16 a, __bf16 b) {
  unsigned short ua = __builtin_bit_cast(unsigned short, a);
  unsigned short ub = __builtin_bit_cast(unsigned short, b);
  return (unsigned)ua | ((unsigned)ub << 16);
}

// --- Kernel 1: embed = embed_sum / clamp(usage, eps); e2[k] = |embed_k|^2 ---
__global__ void prep_embed(const float* __restrict__ embed_sum,
                           const float* __restrict__ usage,
                           float* __restrict__ embed,
                           float* __restrict__ e2) {
  __shared__ float red[256];
  int k = blockIdx.x;
  int t = threadIdx.x;
  float u = usage[k];
  u = (u < EPSV) ? EPSV : u;
  float v = embed_sum[k * DDIM + t] / u;
  embed[k * DDIM + t] = v;
  red[t] = v * v;
  __syncthreads();
  for (int s = 128; s > 0; s >>= 1) {
    if (t < s) red[t] += red[t + s];
    __syncthreads();
  }
  if (t == 0) e2[k] = red[0];
}

// --- Kernel 2: pre-swizzle codebook into WMMA B-operand order (hi/lo bf16) ---
// B (32x16, KxN) 16-bit layout: lane = N + 16*khalf; VGPR j pair base: j<4 -> 2j, j>=4 -> 2j+8, +8 for khalf.
// Per code-tile ct: 4096 dwords = [hi: 8 dt x 32 lanes x 8 dwords][lo: same].
__global__ void prep_pack(const float* __restrict__ embed,
                          unsigned* __restrict__ pack) {
  int blk  = blockIdx.x;        // ct*8 + dt
  int ct   = blk >> 3;
  int dt   = blk & 7;
  int lane = threadIdx.x;       // 0..31
  int n    = ct * 16 + (lane & 15);
  int hf   = lane >> 4;
  const float2* row = (const float2*)(embed + (size_t)n * DDIM);
  unsigned* dst = pack + (size_t)ct * 4096 + dt * 256 + lane * 8;
#pragma unroll
  for (int j = 0; j < 8; ++j) {
    int kl = ((j < 4) ? (2 * j) : (2 * j + 8)) + 8 * hf;  // K-pair base within 32-wide tile
    int d  = dt * 32 + kl;
    float2 f = row[d >> 1];
    __bf16 h0 = (__bf16)f.x, h1 = (__bf16)f.y;
    float  r0 = f.x - (float)h0, r1 = f.y - (float)h1;
    dst[j]        = pack2(h0, h1);
    dst[2048 + j] = pack2((__bf16)r0, (__bf16)r1);
  }
}

#ifdef HAVE_ASYNC_LDS
// Per-thread share of one 16 KB tile = 4 x b128 async copies, coalesced.
static __device__ inline void stage_async(const unsigned* gsrc, unsigned* ldst,
                                          int tid) {
#pragma unroll
  for (int it = 0; it < 4; ++it) {
    const unsigned* g = gsrc + (size_t)(tid + it * 256) * 4;
    unsigned*       l = ldst + (size_t)(tid + it * 256) * 4;
    __builtin_amdgcn_global_load_async_to_lds_b128(
        (AS1 v4i*)(uintptr_t)g,
        (AS3 v4i*)(unsigned)(uintptr_t)l, 0, 0);
  }
}

static __device__ inline void wait_async0() {
#if __has_builtin(__builtin_amdgcn_s_wait_asynccnt)
  __builtin_amdgcn_s_wait_asynccnt(0);
#else
  asm volatile("s_wait_asynccnt 0x0" ::: "memory");
#endif
}
#endif

// 24 WMMAs on one staged code tile: x.e ~= xh.eh + xh.el + xl.eh (f32 accum).
static __device__ inline void tile_compute(const unsigned* lb, int lane,
                                           const v16bf* a_hi, const v16bf* a_lo,
                                           v8f& c) {
#pragma unroll
  for (int dt = 0; dt < NDT; ++dt) {
    v16bf bh = *(const v16bf*)&lb[dt * 256 + lane * 8];
    v16bf bl = *(const v16bf*)&lb[2048 + dt * 256 + lane * 8];
    c = __builtin_amdgcn_wmma_f32_16x16x32_bf16(false, a_hi[dt], false, bh,
                                                (short)0, c, false, false);
    c = __builtin_amdgcn_wmma_f32_16x16x32_bf16(false, a_hi[dt], false, bl,
                                                (short)0, c, false, false);
    c = __builtin_amdgcn_wmma_f32_16x16x32_bf16(false, a_lo[dt], false, bh,
                                                (short)0, c, false, false);
  }
}

// --- Kernel 3: main distance GEMM + argmin + gather ---
// 256 threads = 8 waves; each wave owns a 16-row tile; block covers 128 rows.
__global__ void __launch_bounds__(256) vq_main(const float* __restrict__ x,
                                               const unsigned* __restrict__ pack,
                                               const float* __restrict__ e2,
                                               const float* __restrict__ embed,
                                               int* __restrict__ out_idx,
                                               float* __restrict__ out_q) {
  __shared__ __align__(32) unsigned lds_b[2][4096];  // 2 x 16 KB double buffer
  __shared__ float lds_e2[KCODES];                   // 8 KB
  __shared__ int   lds_idx[128];                     // winners per block

  const int w    = threadIdx.x >> 5;
  const int lane = threadIdx.x & 31;
  const int tile = blockIdx.x * 8 + w;
  const int rowbase = tile * 16;
  const int m  = lane & 15;
  const int hf = lane >> 4;

  // preload e2 into LDS
  for (int i = threadIdx.x; i < KCODES; i += 256) lds_e2[i] = e2[i];

  // Build A fragments (16x32 bf16, A-layout) in VGPRs, hi + lo split.
  // A layout: lane = M + 16*khalf; VGPR j pair base: j<4 -> 2j, j>=4 -> 2j+8, +8 for khalf.
  const float2* xr = (const float2*)(x + (size_t)(rowbase + m) * DDIM);
  v16bf a_hi[NDT], a_lo[NDT];
#pragma unroll
  for (int dt = 0; dt < NDT; ++dt) {
#pragma unroll
    for (int j = 0; j < 8; ++j) {
      int kl = ((j < 4) ? (2 * j) : (2 * j + 8)) + 8 * hf;
      float2 f = xr[(dt * 32 + kl) >> 1];
      __bf16 h0 = (__bf16)f.x, h1 = (__bf16)f.y;
      float  r0 = f.x - (float)h0, r1 = f.y - (float)h1;
      a_hi[dt][2 * j]     = h0;
      a_hi[dt][2 * j + 1] = h1;
      a_lo[dt][2 * j]     = (__bf16)r0;
      a_lo[dt][2 * j + 1] = (__bf16)r1;
    }
  }

  float bestv[8];
  int   besti[8];
#pragma unroll
  for (int r = 0; r < 8; ++r) { bestv[r] = 3.4e38f; besti[r] = 0; }

#ifdef HAVE_ASYNC_LDS
  // Async double-buffered pipeline: tile ct+1 streams into LDS under the
  // WMMAs of tile ct. One s_wait_asynccnt + one barrier per iteration.
  stage_async(pack, &lds_b[0][0], threadIdx.x);
  for (int ct = 0; ct < NCT; ++ct) {
    const int cur = ct & 1;
    wait_async0();        // my async writes into buf[cur] are done
    __syncthreads();      // everyone's writes done; everyone done reading buf[cur^1]
    if (ct + 1 < NCT)
      stage_async(pack + (size_t)(ct + 1) * 4096, &lds_b[cur ^ 1][0],
                  threadIdx.x);
#else
  // Fallback: register-staged double buffer (prefetch tile ct+1 into VGPRs
  // before the barrier; store to the other LDS buffer next iteration).
  uint4 r4[4];
  {
    const uint4* src = (const uint4*)pack;
#pragma unroll
    for (int it = 0; it < 4; ++it) r4[it] = src[threadIdx.x + it * 256];
  }
  for (int ct = 0; ct < NCT; ++ct) {
    const int cur = ct & 1;
    {
      uint4* dst = (uint4*)&lds_b[cur][0];
#pragma unroll
      for (int it = 0; it < 4; ++it) dst[threadIdx.x + it * 256] = r4[it];
    }
    if (ct + 1 < NCT) {
      const uint4* src = (const uint4*)(pack + (size_t)(ct + 1) * 4096);
#pragma unroll
      for (int it = 0; it < 4; ++it) r4[it] = src[threadIdx.x + it * 256];
    }
    __syncthreads();
#endif

    v8f c = {};
    tile_compute(&lds_b[cur][0], lane, a_hi, a_lo, c);

    // score = |e|^2 - 2 x.e   (|x|^2 dropped: constant per row, argmin-invariant)
    float e2v  = lds_e2[ct * 16 + (lane & 15)];
    int   nidx = ct * 16 + (lane & 15);
#pragma unroll
    for (int r = 0; r < 8; ++r) {
      float s = e2v - 2.0f * c[r];
      if (s < bestv[r]) { bestv[r] = s; besti[r] = nidx; }
    }
  }

  // Cross-lane min reduce within each 16-lane half (C layout: row = r + 8*hf),
  // tie-break on lowest index to match jnp.argmin semantics.
#pragma unroll
  for (int r = 0; r < 8; ++r) {
    float bv = bestv[r];
    int   bi = besti[r];
#pragma unroll
    for (int sh = 1; sh < 16; sh <<= 1) {
      float ov = __shfl_xor(bv, sh, 32);
      int   oi = __shfl_xor(bi, sh, 32);
      if (ov < bv || (ov == bv && oi < bi)) { bv = ov; bi = oi; }
    }
    int row = 8 * hf + r;
    if ((lane & 15) == 0) {
      out_idx[rowbase + row] = bi;
      lds_idx[w * 16 + row]  = bi;
    }
  }
  __syncthreads();

  // Gather: quantized[row] = embed[best_idx]; coalesced float4, HBM-bound.
  const float4* eb4 = (const float4*)embed;
  float4* oq4 = (float4*)out_q;
  for (int mm = 0; mm < 16; ++mm) {
    int idx = lds_idx[w * 16 + mm];
    float4 v0 = eb4[(size_t)idx * 64 + lane * 2];
    float4 v1 = eb4[(size_t)idx * 64 + lane * 2 + 1];
    float4* o = oq4 + (size_t)(rowbase + mm) * 64 + lane * 2;
    o[0] = v0;
    o[1] = v1;
  }
}

extern "C" void kernel_launch(void* const* d_in, const int* in_sizes, int n_in,
                              void* d_out, int out_size, void* d_ws, size_t ws_size,
                              hipStream_t stream) {
  (void)n_in; (void)out_size; (void)ws_size;
  const float* x         = (const float*)d_in[0];  // hidden_states [B,T,D]
  const float* embed_sum = (const float*)d_in[1];  // [K,D]
  const float* usage     = (const float*)d_in[2];  // [K]

  float*    embed = (float*)d_ws;
  float*    e2    = embed + (size_t)KCODES * DDIM;
  unsigned* pack  = (unsigned*)(e2 + KCODES);

  const int N = in_sizes[0] / DDIM;                // 65536
  int*   oi = (int*)d_out;                         // indices (int32), N elems
  float* oq = (float*)d_out + N;                   // quantized, N*D f32

  prep_embed<<<KCODES, 256, 0, stream>>>(embed_sum, usage, embed, e2);
  prep_pack<<<NCT * NDT, 32, 0, stream>>>(embed, pack);
  vq_main<<<N / 128, 256, 0, stream>>>(x, pack, e2, embed, oi, oq);
}
